// NodeTimeSeriesDecoder_5162550690547
// MI455X (gfx1250) — matrix operand order repaired and checked by reading, not compiled, not run on hardware
//
#include <hip/hip_runtime.h>

// ---------------- problem constants (from reference) ----------------
#define NODE_DIMS 6
#define GH 128
#define GM 3
#define HD 128
#define TSTEPS 256
#define NODES_PER 2000
#define IN_DIM 137
#define KENC 160            // 137 padded to 5 K-tiles of 32
#define WAVES 8             // waves per workgroup
#define NTOT (32*2000)

typedef __attribute__((ext_vector_type(16))) __bf16 v16bf;
typedef __attribute__((ext_vector_type(8)))  float  v8f;

// ---------------- workspace layout (bf16 elements) ----------------
#define GW_ELEMS   (16*16384)   // 16 chunks: [2 mats][4 gates][16 rows][128 K]
#define WENC_ELEMS (128*KENC)   // WencT [128 out][160 K]
#define WD1_ELEMS  (64*256)     // Wd1   [64 out][256 K]

// ---------------- LDS layout (bytes) ----------------
#define S_WG     0              // 32768  : streamed gate-weight chunk
#define S_WENC   32768          // 40960  : resident encoder weights
#define S_WD1    73728          // 32768  : resident head weights
#define S_BIASL  106496         // 4096   : b_ih+b_hh  [2][512] f32
#define S_BIASE  110592         // 512    : b_enc f32
#define S_BIASD1 111104         // 256    : b_d1 f32
#define S_WD2    111360         // 768    : W_d2 [3][64] f32
#define S_BIASD2 112128         // 16     : b_d2 f32
#define S_WAVE   112640         // per-wave regions start
#define WAVE_BYTES 24576        // bufES 8K | bufX0 4K | bufX1 4K | bufH0 4K | bufH1 4K
#define SMEM_TOTAL (S_WAVE + WAVES*WAVE_BYTES)   // 309248 B  (< 320 KB WGP LDS)

// ---------------- fragment helpers ----------------
union FragU { v16bf bf; uint4 u[2]; };
union PkU   { __bf16 b[2]; unsigned u; };

// A fragment: 16x32 bf16, row-major LDS buffer, row stride ldk (elements, 16B aligned)
__device__ __forceinline__ v16bf load_fragA(const __bf16* __restrict__ buf,
                                            int ldk, int kbase, int m, int half) {
  FragU f;
  f.u[0] = *(const uint4*)(buf + m*ldk + kbase + half*8);        // VGPR0-3: K=kb..+7 / +8..15
  f.u[1] = *(const uint4*)(buf + m*ldk + kbase + 16 + half*8);   // VGPR4-7: K=+16..23 / +24..31
  return f.bf;
}
// B fragment: 32x16 bf16 from weight stored [out][K] row-major; lane = out column
__device__ __forceinline__ v16bf load_fragB(const __bf16* __restrict__ w,
                                            int ldk, int kbase, int m, int half) {
  FragU f;
  f.u[0] = *(const uint4*)(w + m*ldk + kbase + half*8);          // K = kb..kb+7  (or +8..15)
  f.u[1] = *(const uint4*)(w + m*ldk + kbase + 16 + half*8);
  return f.bf;
}

__device__ __forceinline__ v8f wmma_bf16(v16bf a, v16bf b, v8f c) {
  return __builtin_amdgcn_wmma_f32_16x16x32_bf16(false, a, false, b, (short)0, c,
                                                 false, false);
}
__device__ __forceinline__ float sigm(float x) { return 1.0f / (1.0f + __expf(-x)); }
__device__ __forceinline__ float tanh_fast(float x) {
  x = fminf(15.0f, fmaxf(-15.0f, x));
  float e = __expf(2.0f * x);
  return (e - 1.0f) / (e + 1.0f);
}
__device__ __forceinline__ v8f splat8(float v) {
  v8f r;
#pragma unroll
  for (int i = 0; i < 8; ++i) r[i] = v;
  return r;
}

// ---------------- prep kernels: fp32 weights -> bf16, reordered ----------------
__global__ void k_prep_gates(const float* __restrict__ W_ih,
                             const float* __restrict__ W_hh,
                             __bf16* __restrict__ gw) {
  int i = blockIdx.x * 256 + threadIdx.x;
  if (i >= GW_ELEMS) return;
  int c = i >> 14, e = i & 16383;          // chunk, elem
  int l = c >> 3, j = c & 7;               // layer, column-tile
  int mat = e >> 13, rem = e & 8191;       // 0=ih 1=hh
  int g = rem >> 11;                       // gate i,f,g,o
  int r = (rem >> 7) & 15;                 // row within 16-wide tile
  int k = rem & 127;
  const float* src = mat ? W_hh : W_ih;
  int row = g * 128 + j * 16 + r;          // row in [512]
  gw[i] = (__bf16)src[(l * 512 + row) * 128 + k];
}
__global__ void k_prep_enc(const float* __restrict__ W_enc, __bf16* __restrict__ wencT) {
  int i = blockIdx.x * 256 + threadIdx.x;
  if (i >= WENC_ELEMS) return;
  int j = i / KENC, k = i % KENC;
  wencT[i] = (__bf16)((k < IN_DIM) ? W_enc[j * IN_DIM + k] : 0.0f);
}
__global__ void k_prep_d1(const float* __restrict__ W_d1, __bf16* __restrict__ wd1) {
  int i = blockIdx.x * 256 + threadIdx.x;
  if (i >= WD1_ELEMS) return;
  wd1[i] = (__bf16)W_d1[i];
}

// ---------------- main persistent-time-loop kernel ----------------
__global__ __launch_bounds__(256)
void k_lstm_decode(const float* __restrict__ node,
                   const float* __restrict__ latent,   // [32][256][128]
                   const float* __restrict__ gms,      // [32][256][3]
                   const float* __restrict__ b_enc,
                   const float* __restrict__ b_ih,     // [2][512]
                   const float* __restrict__ b_hh,
                   const float* __restrict__ b_d1,
                   const float* __restrict__ W_d2,     // [3][64]
                   const float* __restrict__ b_d2,
                   const __bf16* __restrict__ gw,
                   const __bf16* __restrict__ wencT,
                   const __bf16* __restrict__ wd1,
                   float* __restrict__ out)            // [N][256][3]
{
  extern __shared__ char smem[];
  __bf16* sWg     = (__bf16*)(smem + S_WG);
  __bf16* sWencT  = (__bf16*)(smem + S_WENC);
  __bf16* sWd1    = (__bf16*)(smem + S_WD1);
  float*  sBiasL  = (float*)(smem + S_BIASL);
  float*  sBiasE  = (float*)(smem + S_BIASE);
  float*  sBiasD1 = (float*)(smem + S_BIASD1);
  float*  sWd2    = (float*)(smem + S_WD2);
  float*  sBiasD2 = (float*)(smem + S_BIASD2);

  const int tid  = threadIdx.x;
  const int wave = tid >> 5, lane = tid & 31;
  const int m = lane & 15, half = lane >> 4;

  char*   my    = smem + S_WAVE + wave * WAVE_BYTES;
  __bf16* bufES = (__bf16*)my;             // encode in [16][160] / state [16][256]
  __bf16* bufX0 = (__bf16*)(my + 8192);    // layer0 input x        [16][128]
  __bf16* bufX1 = (__bf16*)(my + 12288);   // layer1 input          [16][128]
  __bf16* bufH0 = (__bf16*)(my + 16384);   // h layer0              [16][128]
  __bf16* bufH1 = (__bf16*)(my + 20480);   // h layer1              [16][128]
  float*  bufR  = (float*)bufX0;           // relu output f32 [16][64] (x0 dead)

  // ---- resident LDS preload (once) ----
  for (int i = tid; i < WENC_ELEMS / 8; i += 256)
    ((uint4*)sWencT)[i] = ((const uint4*)wencT)[i];
  for (int i = tid; i < WD1_ELEMS / 8; i += 256)
    ((uint4*)sWd1)[i] = ((const uint4*)wd1)[i];
  for (int i = tid; i < 1024; i += 256) sBiasL[i] = b_ih[i] + b_hh[i];
  if (tid < 128) sBiasE[tid]  = b_enc[tid];
  if (tid < 64)  sBiasD1[tid] = b_d1[tid];
  if (tid < 192) sWd2[tid]    = W_d2[tid];
  if (tid < 3)   sBiasD2[tid] = b_d2[tid];

  const int tileIdx = blockIdx.x * WAVES + wave;     // 16-node tile per wave
  const int node0   = tileIdx * 16;
  const int bg      = node0 / NODES_PER;             // graph id (tile-uniform)

  // static node-feature columns + zero padding (written once)
  if (half == 0) {
#pragma unroll
    for (int c = 0; c < NODE_DIMS; ++c)
      bufES[m * KENC + c] = (__bf16)node[(size_t)(node0 + m) * NODE_DIMS + c];
#pragma unroll
    for (int c = IN_DIM; c < KENC; ++c) bufES[m * KENC + c] = (__bf16)0.0f;
  }
  __syncthreads();

  v8f c0[8], c1[8];                        // persistent cell state, VGPR-resident
  __bf16 gm0 = (__bf16)0.0f, gm1 = (__bf16)0.0f, gm2 = (__bf16)0.0f;

  // stage dynamic encode columns for step tt and run the encoder GEMM.
  // `init` is a call-site constant: folded after inlining (no runtime branch).
  auto stage_encode = [&](int tt, bool init) {
    const float* latrow = latent + ((size_t)bg * TSTEPS + tt) * GH;
    const float* gmrow  = gms    + ((size_t)bg * TSTEPS + tt) * GM;
    // latent cols 6..133 as 64 packed bf16 pairs -> ds_store_b32 broadcasts
    for (int p = lane; p < GH / 2; p += 32) {
      PkU pk;
      pk.b[0] = (__bf16)latrow[2 * p];
      pk.b[1] = (__bf16)latrow[2 * p + 1];
      int col = NODE_DIMS + 2 * p;                   // even -> 4B aligned
#pragma unroll
      for (int r = 0; r < 16; ++r)
        *(unsigned*)(bufES + r * KENC + col) = pk.u;
    }
    if (lane < GM) {                                 // gms cols 134..136
      __bf16 bv = (__bf16)gmrow[lane];
#pragma unroll
      for (int r = 0; r < 16; ++r) bufES[r * KENC + NODE_DIMS + GH + lane] = bv;
    }
    gm0 = (__bf16)gmrow[0]; gm1 = (__bf16)gmrow[1]; gm2 = (__bf16)gmrow[2];

    // encoder GEMM: x[16x128] = E[16x160] @ WencT, 5 K-tiles
    v16bf ae[5];
#pragma unroll
    for (int kt = 0; kt < 5; ++kt) ae[kt] = load_fragA(bufES, KENC, kt * 32, m, half);
#pragma unroll
    for (int jc = 0; jc < 8; ++jc) {
      v8f acc = splat8(sBiasE[jc * 16 + m]);
#pragma unroll
      for (int kt = 0; kt < 5; ++kt)
        acc = wmma_bf16(ae[kt], load_fragB(sWencT + (jc * 16) * KENC, KENC, kt * 32, m, half), acc);
#pragma unroll
      for (int v = 0; v < 8; ++v) {
        int row = v + half * 8;
        __bf16 xb = (__bf16)acc[v];
        bufX0[row * HD + jc * 16 + m] = xb;
        if (init) {                                  // H0 = C0 = x0, both layers
          bufH0[row * HD + jc * 16 + m] = xb;
          bufH1[row * HD + jc * 16 + m] = xb;
        }
      }
      if (init) { c0[jc] = acc; c1[jc] = acc; }
    }
  };

  // one LSTM layer; `l` is a call-site constant and `cs` binds directly to
  // c0/c1 by reference so the cell state stays in VGPRs (no scratch demotion).
  auto lstm_layer = [&](const int l, __bf16* __restrict__ Xb,
                        __bf16* __restrict__ Hb, v8f (&cs)[8]) {
#pragma unroll 1
    for (int j = 0; j < 8; ++j) {
      __syncthreads();                            // previous chunk consumed
      const uint4* src = (const uint4*)(gw + (size_t)(l * 8 + j) * 16384);
      for (int i = tid; i < 2048; i += 256) ((uint4*)sWg)[i] = src[i];
      __syncthreads();
      int nc = l * 8 + j + 1;                     // prefetch next chunk into L2
      if (nc < 16) __builtin_prefetch(gw + (size_t)nc * 16384 + tid * 64, 0, 1);

      v16bf ax[4], ah[4];
#pragma unroll
      for (int kt = 0; kt < 4; ++kt) {
        ax[kt] = load_fragA(Xb, HD, kt * 32, m, half);
        ah[kt] = load_fragA(Hb, HD, kt * 32, m, half);
      }
      v8f g4[4];
#pragma unroll
      for (int g = 0; g < 4; ++g) {
        v8f acc = splat8(sBiasL[l * 512 + g * 128 + j * 16 + m]);
        const __bf16* wih = sWg + (g * 16) * HD;            // [mat0][gate g]
        const __bf16* whh = sWg + 8192 + (g * 16) * HD;     // [mat1][gate g]
#pragma unroll
        for (int kt = 0; kt < 4; ++kt) {
          acc = wmma_bf16(ax[kt], load_fragB(wih, HD, kt * 32, m, half), acc);
          acc = wmma_bf16(ah[kt], load_fragB(whh, HD, kt * 32, m, half), acc);
        }
        g4[g] = acc;
      }
      v8f c2, h2;
#pragma unroll
      for (int v = 0; v < 8; ++v) {
        float ig = sigm(g4[0][v]), fg = sigm(g4[1][v]);
        float gg = tanh_fast(g4[2][v]), og = sigm(g4[3][v]);
        float cv = fg * cs[j][v] + ig * gg;
        c2[v] = cv;
        h2[v] = og * tanh_fast(cv);
      }
      cs[j] = c2;
#pragma unroll
      for (int v = 0; v < 8; ++v) {
        int row = v + half * 8, col = j * 16 + m;
        __bf16 hb = (__bf16)h2[v];
        Hb[row * HD + col] = hb;
        if (l == 0) {                 // layer1 input: h0 with cols 125..127 <- gms
          __bf16 xv = hb;
          if (col == 125) xv = gm0; else if (col == 126) xv = gm1; else if (col == 127) xv = gm2;
          bufX1[row * HD + col] = xv;
        } else {                      // state s = [h1 | c1] for the head
          bufES[row * 256 + col]       = hb;
          bufES[row * 256 + 128 + col] = (__bf16)c2[v];
        }
      }
    }
  };

  stage_encode(0, true);                             // peeled init (t = 0 state)

  const float bd0 = sBiasD2[0], bd1v = sBiasD2[1], bd2v = sBiasD2[2];
  float* outp = out + (size_t)(node0 + m) * TSTEPS * 3;   // lanes 0..15 write

#pragma unroll 1
  for (int t = 0; t < TSTEPS; ++t) {
    // ---- two LSTM layers, gate weights streamed per 16-col tile ----
    lstm_layer(0, bufX0, bufH0, c0);
    lstm_layer(1, bufX1, bufH1, c1);

    // ---- head: r = relu(s @ Wd1T + b_d1), 16x64 ----
    v16bf as[8];
#pragma unroll
    for (int kt = 0; kt < 8; ++kt) as[kt] = load_fragA(bufES, 256, kt * 32, m, half);
#pragma unroll
    for (int jc = 0; jc < 4; ++jc) {
      v8f acc = splat8(sBiasD1[jc * 16 + m]);
#pragma unroll
      for (int kt = 0; kt < 8; ++kt)
        acc = wmma_bf16(as[kt], load_fragB(sWd1 + (jc * 16) * 256, 256, kt * 32, m, half), acc);
#pragma unroll
      for (int v = 0; v < 8; ++v) {
        int row = v + half * 8;
        bufR[row * 64 + jc * 16 + m] = fmaxf(acc[v], 0.0f);
      }
    }
    // ---- y = r @ Wd2T + b_d2 (tiny, VALU), lanes 0..15 = nodes ----
    if (half == 0) {
      float y0 = bd0, y1 = bd1v, y2 = bd2v;
#pragma unroll 8
      for (int c = 0; c < 64; ++c) {
        float rv = bufR[m * 64 + c];
        y0 = fmaf(rv, sWd2[c], y0);
        y1 = fmaf(rv, sWd2[64 + c], y1);
        y2 = fmaf(rv, sWd2[128 + c], y2);
      }
      outp[0] = y0; outp[1] = y1; outp[2] = y2;
      outp += 3;
    }

    // ---- stage + encode for the NEXT step (keeps hot loop branch-free) ----
    if (t + 1 < TSTEPS) stage_encode(t + 1, false);
  }
}

// ---------------- launch ----------------
extern "C" void kernel_launch(void* const* d_in, const int* in_sizes, int n_in,
                              void* d_out, int out_size, void* d_ws, size_t ws_size,
                              hipStream_t stream) {
  const float* node   = (const float*)d_in[0];
  // d_in[1] = ptr (uniform 2000-node graphs; batch id derived arithmetically)
  const float* latent = (const float*)d_in[2];
  const float* gmsp   = (const float*)d_in[3];
  const float* W_enc  = (const float*)d_in[4];
  const float* b_enc  = (const float*)d_in[5];
  const float* W_ih   = (const float*)d_in[6];
  const float* W_hh   = (const float*)d_in[7];
  const float* b_ih   = (const float*)d_in[8];
  const float* b_hh   = (const float*)d_in[9];
  const float* W_d1   = (const float*)d_in[10];
  const float* b_d1   = (const float*)d_in[11];
  const float* W_d2   = (const float*)d_in[12];
  const float* b_d2   = (const float*)d_in[13];
  float* out = (float*)d_out;

  __bf16* gw    = (__bf16*)d_ws;
  __bf16* wencT = gw + GW_ELEMS;
  __bf16* wd1   = wencT + WENC_ELEMS;

  k_prep_gates<<<(GW_ELEMS + 255) / 256, 256, 0, stream>>>(W_ih, W_hh, gw);
  k_prep_enc<<<(WENC_ELEMS + 255) / 256, 256, 0, stream>>>(W_enc, wencT);
  k_prep_d1<<<(WD1_ELEMS + 255) / 256, 256, 0, stream>>>(W_d1, wd1);

  int nblocks = NTOT / (WAVES * 16);   // 500 workgroups x 8 wave-tiles x 16 nodes
  k_lstm_decode<<<nblocks, 256, SMEM_TOTAL, stream>>>(
      node, latent, gmsp, b_enc, b_ih, b_hh, b_d1, W_d2, b_d2,
      gw, wencT, wd1, out);
}
